// Lost_19542101197056
// MI455X (gfx1250) — matrix operand order repaired and checked by reading, not compiled, not run on hardware
//
#include <hip/hip_runtime.h>
#include <hip/hip_bf16.h>

// ---------------------------------------------------------------------------
// LOST-style seed segmentation for MI455X (gfx1250, wave32, WMMA).
//   sim = out @ out.T  -> v_wmma_f32_16x16x32_bf16, 128x128 tiles,
//   double-buffered LDS fed by global_load_async_to_lds_b128 (ASYNCcnt).
//   sim (64MB) stays resident in the 192MB L2 for all downstream passes.
// ---------------------------------------------------------------------------

#define NPATCH 4096
#define DIM    768
#define DWROW  (DIM/2)      // dwords per bf16 row = 384
#define KTILES (DIM/32)     // 24
#define HD     64
#define WD     64
#define KSEL   100
#define BIGL   (NPATCH + 1)
#define LDSPITCH 20         // 16 dwords payload + 4 pad -> 80B rows (16B aligned)

typedef __attribute__((ext_vector_type(16))) __bf16       v16bf;
typedef __attribute__((ext_vector_type(8)))  float        v8f;
typedef __attribute__((ext_vector_type(8)))  unsigned int v8u;
typedef int v4i __attribute__((vector_size(16)));

#if __has_builtin(__builtin_amdgcn_global_load_async_to_lds_b128)
#define HAS_ASYNC_LDS 1
#else
#define HAS_ASYNC_LDS 0
#endif

__device__ __forceinline__ void async_cp16(const unsigned int* g, unsigned int* l) {
#if HAS_ASYNC_LDS
    __builtin_amdgcn_global_load_async_to_lds_b128(
        (__attribute__((address_space(1))) v4i*)g,
        (__attribute__((address_space(3))) v4i*)l,
        0, 0);
#else
    *(uint4*)l = *(const uint4*)g;   // fallback: global_load_b128 + ds_store_b128
#endif
}

template <int N>
__device__ __forceinline__ void wait_async() {
#if HAS_ASYNC_LDS
#if __has_builtin(__builtin_amdgcn_s_wait_asynccnt)
    __builtin_amdgcn_s_wait_asynccnt(N);
#else
    asm volatile("s_wait_asynccnt %0" :: "i"(N) : "memory");
#endif
#endif
}

__device__ __forceinline__ unsigned short f32_to_bf16_rne(float f) {
    unsigned int u = __float_as_uint(f);
    u += 0x7FFFu + ((u >> 16) & 1u);          // round-to-nearest-even
    return (unsigned short)(u >> 16);
}

// ---- 1) f32 -> bf16 conversion of the feature matrix -----------------------
__global__ __launch_bounds__(256) void k_cvt(const float* __restrict__ src,
                                             unsigned short* __restrict__ dst,
                                             int n) {
    int i = blockIdx.x * blockDim.x + threadIdx.x;
    if (i < n) dst[i] = f32_to_bf16_rne(src[i]);
}

// ---- 2) sim = out @ out.T via WMMA bf16 -----------------------------------
// 256 threads = 8 waves; block tile 128x128; wave w owns rows [16w,16w+16),
// 8 accumulator tiles (16x16) across 128 columns; K pipeline 24 x k=32 with
// double-buffered async global->LDS staging.
__device__ __forceinline__ void stage_tiles(const unsigned int* __restrict__ B16,
                                            unsigned int (*As)[LDSPITCH],
                                            unsigned int (*Bs)[LDSPITCH],
                                            int rowBase, int colBase, int kd, int tid) {
    // 128x16 dwords per tile = 512 x 16B chunks; 256 threads x 2 chunks each
    #pragma unroll
    for (int j = 0; j < 2; ++j) {
        int e  = tid + 256 * j;          // 0..511
        int r  = e >> 2;                 // row 0..127
        int c4 = (e & 3) << 2;           // dword col 0/4/8/12
        async_cp16(&B16[(size_t)(rowBase + r) * DWROW + kd + c4], &As[r][c4]);
        async_cp16(&B16[(size_t)(colBase + r) * DWROW + kd + c4], &Bs[r][c4]);
    }
}

__global__ __launch_bounds__(256) void k_gemm(const unsigned int* __restrict__ B16,
                                              float* __restrict__ sim) {
    __shared__ __align__(16) unsigned int As[2][128][LDSPITCH];
    __shared__ __align__(16) unsigned int Bs[2][128][LDSPITCH];

    const int tid  = threadIdx.x;
    const int lane = tid & 31;
    const int wave = tid >> 5;
    const int half = lane >> 4;
    const int l15  = lane & 15;
    const int rowBase = blockIdx.y * 128;
    const int colBase = blockIdx.x * 128;

    v8f acc[8] = {};

    stage_tiles(B16, As[0], Bs[0], rowBase, colBase, 0, tid);

    for (int kt = 0; kt < KTILES; ++kt) {
        const int cur = kt & 1;
        if (kt + 1 < KTILES) {
            stage_tiles(B16, As[cur ^ 1], Bs[cur ^ 1], rowBase, colBase,
                        (kt + 1) * 16, tid);
            wait_async<4>();             // current tile's 4 ops done; next 4 in flight
        } else {
            wait_async<0>();
        }
        __syncthreads();                 // all waves' LDS writes visible

        // A fragment (16x32 bf16): VGPR v<4 -> K=2v+8*half, v>=4 -> +16
        v8u au;
        const int am = wave * 16 + l15;
        #pragma unroll
        for (int v = 0; v < 8; ++v) {
            int dw = (v & 3) + 4 * half + 8 * (v >> 2);
            au[v] = As[cur][am][dw];
        }
        v16bf afrag = __builtin_bit_cast(v16bf, au);

        #pragma unroll
        for (int t = 0; t < 8; ++t) {
            // B fragment (32x16 bf16): VGPR v<4 -> K=2v+8*half, v>=4 -> +16
            v8u bu;
            const int bn = t * 16 + l15;
            #pragma unroll
            for (int v = 0; v < 8; ++v) bu[v] = Bs[cur][bn][(v & 3) + 4 * half + 8 * (v >> 2)];
            v16bf bfrag = __builtin_bit_cast(v16bf, bu);
            acc[t] = __builtin_amdgcn_wmma_f32_16x16x32_bf16(
                false, afrag, false, bfrag, (short)0, acc[t], false, false);
        }
        __syncthreads();                 // reads done before buffer is re-filled
    }

    // C layout: lanes 0-15 -> M=r, lanes 16-31 -> M=8+r; N = lane&15
    const int gr0 = rowBase + wave * 16 + 8 * half;
    #pragma unroll
    for (int t = 0; t < 8; ++t) {
        const int gc = colBase + t * 16 + l15;
        #pragma unroll
        for (int r = 0; r < 8; ++r)
            sim[(size_t)(gr0 + r) * NPATCH + gc] = acc[t][r];
    }
}

// ---- 3) per-row nonneg counts (sim symmetric => col sums == row sums) ------
__global__ __launch_bounds__(256) void k_rowcnt(const float* __restrict__ sim,
                                                int* __restrict__ cnt) {
    __shared__ int red[256];
    const float* p = sim + (size_t)blockIdx.x * NPATCH;
    int c = 0;
    for (int j = threadIdx.x; j < NPATCH; j += 256) c += (p[j] >= 0.0f);
    red[threadIdx.x] = c;
    __syncthreads();
    for (int s = 128; s > 0; s >>= 1) {
        if (threadIdx.x < s) red[threadIdx.x] += red[threadIdx.x + s];
        __syncthreads();
    }
    if (threadIdx.x == 0) cnt[blockIdx.x] = red[0];
}

// ---- 4) argmin (first-min tie rule) + seed blend ---------------------------
__global__ __launch_bounds__(256) void k_seed(const int* __restrict__ cnt,
                                              const int* __restrict__ attn,
                                              int* __restrict__ seedbuf) {
    __shared__ int bv[256], bi[256];
    int bestv = 0x7fffffff, besti = 0;
    for (int j = threadIdx.x; j < NPATCH; j += 256) {
        int v = cnt[j];
        if (v < bestv) { bestv = v; besti = j; }
    }
    bv[threadIdx.x] = bestv; bi[threadIdx.x] = besti;
    __syncthreads();
    for (int s = 128; s > 0; s >>= 1) {
        if (threadIdx.x < s) {
            int v2 = bv[threadIdx.x + s], i2 = bi[threadIdx.x + s];
            if (v2 < bv[threadIdx.x] ||
                (v2 == bv[threadIdx.x] && i2 < bi[threadIdx.x])) {
                bv[threadIdx.x] = v2; bi[threadIdx.x] = i2;
            }
        }
        __syncthreads();
    }
    if (threadIdx.x == 0) {
        int f = bi[0];
        int r = f / WD, c = f % WD;
        // jnp.round is round-half-to-even, same as rintf default mode
        int sr = (int)rintf(0.5f * (float)(r + attn[0]));
        int sc = (int)rintf(0.5f * (float)(c + attn[1]));
        seedbuf[0] = sr * WD + sc;
    }
}

// ---- 5) s init: deg[seed] row, s[seed]=1, keep stable-rank < K -------------
__global__ __launch_bounds__(1024) void k_inits(const float* __restrict__ sim,
                                                const int* __restrict__ seedbuf,
                                                int* __restrict__ sArr) {
    __shared__ float rowv[NPATCH];
    const int seed = seedbuf[0];
    const float* p = sim + (size_t)seed * NPATCH;
    for (int j = threadIdx.x; j < NPATCH; j += 1024) rowv[j] = p[j];
    __syncthreads();
    for (int j = threadIdx.x; j < NPATCH; j += 1024) {
        float v = rowv[j];
        int gt = 0, eq = 0;
        for (int t = 0; t < NPATCH; ++t) {
            float w = rowv[t];
            gt += (w > v);
            eq += (w == v) & (t < j);      // stable argsort(-x) tie rule
        }
        int s = (v >= 0.0f) ? 1 : 0;
        if (j == seed) s = 1;
        if (gt + eq >= KSEL) s = 0;
        sArr[j] = s;
    }
}

// ---- 6) sequential expansion (restricted to the <=K active set, which is
//         exact: s[i]==0 entries can never change) + connected components ----
__global__ __launch_bounds__(1024) void k_scan_cc(const float* __restrict__ sim,
                                                  const int* __restrict__ sArr,
                                                  float* __restrict__ outp) {
    __shared__ int   sl[NPATCH];
    __shared__ int   ones[128];
    __shared__ float part[128];
    __shared__ int   cntS;
    __shared__ int   lab[NPATCH];
    __shared__ int   chg;
    const int tid = threadIdx.x;

    for (int j = tid; j < NPATCH; j += 1024) sl[j] = sArr[j];
    __syncthreads();
    if (tid == 0) {
        int c = 0;
        for (int j = 0; j < NPATCH && c < 128; ++j)
            if (sl[j]) ones[c++] = j;
        cntS = c;
    }
    __syncthreads();
    const int cnt = cntS;

    // sequential scan over active indices in ascending order
    for (int o = 0; o < cnt; ++o) {
        const int i = ones[o];
        if (tid < 128) {
            float v = 0.0f;
            if (tid < cnt) {
                int k = ones[tid];
                if (sl[k]) v = sim[(size_t)i * NPATCH + k];   // L2 hit
            }
            part[tid] = v;
        }
        __syncthreads();
        if (tid == 0) {
            float sum = 0.0f;
            for (int t = 0; t < cnt; ++t) sum += part[t];
            sl[i] = (sum > 0.0f) ? 1 : 0;
        }
        __syncthreads();
    }

    // 8-connectivity connected components: in-place min-label propagation
    for (int j = tid; j < NPATCH; j += 1024) lab[j] = sl[j] ? (j + 1) : BIGL;
    __syncthreads();
    for (int it = 0; it < 8192; ++it) {
        if (tid == 0) chg = 0;
        __syncthreads();
        for (int j = tid; j < NPATCH; j += 1024) {
            if (sl[j]) {
                int y = j >> 6, x = j & 63;
                int m = lab[j];
                #pragma unroll
                for (int dy = -1; dy <= 1; ++dy)
                    #pragma unroll
                    for (int dx = -1; dx <= 1; ++dx) {
                        int ny = y + dy, nx = x + dx;
                        if (ny >= 0 && ny < HD && nx >= 0 && nx < WD) {
                            int lv = lab[(ny << 6) + nx];
                            m = lv < m ? lv : m;
                        }
                    }
                if (m < lab[j]) { lab[j] = m; chg = 1; }
            }
        }
        __syncthreads();
        int done = (chg == 0);
        __syncthreads();
        if (done) break;
    }
    __syncthreads();
    for (int j = tid; j < NPATCH; j += 1024)
        outp[j] = sl[j] ? (float)lab[j] : 0.0f;
}

// ---------------------------------------------------------------------------
extern "C" void kernel_launch(void* const* d_in, const int* in_sizes, int n_in,
                              void* d_out, int out_size, void* d_ws, size_t ws_size,
                              hipStream_t stream) {
    const float* outF = (const float*)d_in[0];   // (N, D) f32
    const int*   attn = (const int*)d_in[1];     // (2,) int32

    char* ws = (char*)d_ws;
    unsigned short* outb    = (unsigned short*)ws;                      // 6 MB bf16
    float*          sim     = (float*)(ws + ((size_t)8  << 20));        // 64 MB f32
    int*            cnt     = (int*)  (ws + ((size_t)72 << 20));        // 16 KB
    int*            seedbuf = (int*)  (ws + ((size_t)72 << 20) + 16384);
    int*            sArr    = (int*)  (ws + ((size_t)72 << 20) + 16384 + 256);

    k_cvt<<<(NPATCH * DIM + 255) / 256, 256, 0, stream>>>(outF, outb, NPATCH * DIM);

    dim3 grid(NPATCH / 128, NPATCH / 128);       // 32 x 32 blocks
    k_gemm<<<grid, 256, 0, stream>>>((const unsigned int*)outb, sim);

    k_rowcnt<<<NPATCH, 256, 0, stream>>>(sim, cnt);
    k_seed  <<<1, 256, 0, stream>>>(cnt, attn, seedbuf);
    k_inits <<<1, 1024, 0, stream>>>(sim, seedbuf, sArr);
    k_scan_cc<<<1, 1024, 0, stream>>>(sim, sArr, (float*)d_out);
}
